// ImprintedModel_73735998537873
// MI455X (gfx1250) — matrix operand order repaired and checked by reading, not compiled, not run on hardware
//
#include <hip/hip_runtime.h>
#include <stdint.h>

// Problem constants (match reference)
#define B_N   4096
#define E_K   512
#define C_CLS 2000
#define PROX  32
#define P_ROWS (C_CLS * PROX)   // 64000

typedef __attribute__((ext_vector_type(16))) __bf16 v16bf;
typedef __attribute__((ext_vector_type(8)))  float  v8f;

static __device__ __forceinline__ uint16_t f32_to_bf16(float f) {
    uint32_t u = __float_as_uint(f);
    u += 0x7FFFu + ((u >> 16) & 1u);   // round-to-nearest-even
    return (uint16_t)(u >> 16);
}

// CDNA5 async global->LDS copy (VGLOBAL encoding, tracked by ASYNCcnt).
// ldsoff = low 32 bits of generic shared pointer == addrspace(3) offset.
static __device__ __forceinline__ void async_copy_b128(uint32_t ldsoff, const void* gsrc) {
    asm volatile("global_load_async_to_lds_b128 %0, %1, off"
                 :: "v"(ldsoff), "v"(gsrc) : "memory");
}
static __device__ __forceinline__ void wait_asynccnt0() {
    asm volatile("s_wait_asynccnt 0x0" ::: "memory");
}

// ---------------------------------------------------------------------------
// Stage 1: L2-normalize each 512-wide row, emit bf16. One 256-thread block/row.
// ---------------------------------------------------------------------------
__global__ void __launch_bounds__(256)
norm_bf16_kernel(const float* __restrict__ src, uint16_t* __restrict__ dst) {
    const int row = blockIdx.x;
    const int tid = threadIdx.x;              // 0..255
    const float* r = src + (size_t)row * E_K;
    float x0 = r[tid];
    float x1 = r[tid + 256];
    float s  = x0 * x0 + x1 * x1;
    #pragma unroll
    for (int m = 16; m >= 1; m >>= 1) s += __shfl_xor(s, m, 32);
    __shared__ float red[8];
    if ((tid & 31) == 0) red[tid >> 5] = s;
    __syncthreads();
    float tot = 0.f;
    #pragma unroll
    for (int i = 0; i < 8; ++i) tot += red[i];
    const float inv = 1.0f / fmaxf(sqrtf(tot), 1e-12f);
    uint16_t* o = dst + (size_t)row * E_K;
    o[tid]       = f32_to_bf16(x0 * inv);
    o[tid + 256] = f32_to_bf16(x1 * inv);
}

// ---------------------------------------------------------------------------
// Stage 2: fused (class-block GEMM + per-class max), double-buffered LDS with
// async global->LDS copies overlapping WMMA compute.
// Block = 256 threads = 8 waves. blockIdx.y = class, blockIdx.x = B/256 tile.
// ---------------------------------------------------------------------------
#define LSTR   72      // padded LDS row stride in bf16: 144 B (16B-aligned, bank-safe)
#define KCH    64      // K chunk per LDS stage
#define NCHUNK (E_K / KCH)   // 8

__global__ void __launch_bounds__(256)
proxy_gemm_max_kernel(const uint16_t* __restrict__ wn,   // [P_ROWS, E_K] bf16 (normalized)
                      const uint16_t* __restrict__ dn,   // [B_N,    E_K] bf16 (normalized)
                      float* __restrict__ out) {         // [C_CLS, B_N]
    __shared__ uint16_t w_lds[2][32  * LSTR];   //  2 x 4.5 KB
    __shared__ uint16_t d_lds[2][256 * LSTR];   //  2 x 36  KB
    __shared__ float    part[2][256];           //  2 KB      (~83.5 KB total)

    const int nb    = blockIdx.x;            // 0..15
    const int cls   = blockIdx.y;            // 0..1999
    const int tid   = threadIdx.x;
    const int lane  = tid & 31;
    const int wv    = tid >> 5;              // 0..7
    const int mh    = wv & 1;                // M half
    const int nq    = wv >> 1;               // N quarter
    const int bbase = nb * 256;

    const int m    = lane & 15;              // M row / N col within 16x16 tile
    const int half = lane >> 4;              // lane half (ISA fragment layout)

    const int wr   = tid >> 3;               // 0..31 (A staging row)
    const int wseg = tid & 7;                // 16B segment

    v8f acc[4] = {v8f{}, v8f{}, v8f{}, v8f{}};

    // Issue one chunk's async copies (1x A + 8x B uint4 per thread).
    auto prefetch = [&](int ko, int bufi) {
        async_copy_b128((uint32_t)(uintptr_t)&w_lds[bufi][wr * LSTR + wseg * 8],
                        wn + ((size_t)(cls * 32 + wr) * E_K + ko + wseg * 8));
        #pragma unroll
        for (int i = 0; i < 8; ++i) {
            const int idx = i * 256 + tid;
            const int r   = idx >> 3;
            const int seg = idx & 7;
            async_copy_b128((uint32_t)(uintptr_t)&d_lds[bufi][r * LSTR + seg * 8],
                            dn + ((size_t)(bbase + r) * E_K + ko + seg * 8));
        }
    };

    prefetch(0, 0);

    for (int c = 0; c < NCHUNK; ++c) {
        wait_asynccnt0();        // my wave's async LDS writes for chunk c done
        __syncthreads();         // ... and every other wave's too

        if (c + 1 < NCHUNK) prefetch((c + 1) * KCH, (c + 1) & 1);  // overlaps compute

        const uint16_t* wb = &w_lds[c & 1][0];
        const uint16_t* db = &d_lds[c & 1][0];

        #pragma unroll
        for (int kk = 0; kk < KCH; kk += 32) {
            // A fragment: 16x32 bf16, contiguous K layout (lane half selects K 0-15 / 16-31)
            union { uint32_t u[8]; v16bf v; } a;
            const uint16_t* ab = wb + (mh * 16 + m) * LSTR + kk;
            #pragma unroll
            for (int j = 0; j < 8; ++j) {
                const int koff = 2 * j + half * 16;
                a.u[j] = *(const uint32_t*)(ab + koff);
            }
            // Batch all 4 B fragments, then issue 4 back-to-back WMMAs.
            union { uint32_t u[8]; v16bf v; } b[4];
            #pragma unroll
            for (int t = 0; t < 4; ++t) {
                const uint16_t* bp = db + (nq * 64 + t * 16 + m) * LSTR + kk + half * 16;
                #pragma unroll
                for (int j = 0; j < 8; ++j)
                    b[t].u[j] = *(const uint32_t*)(bp + 2 * j);
            }
            #pragma unroll
            for (int t = 0; t < 4; ++t)
                acc[t] = __builtin_amdgcn_wmma_f32_16x16x32_bf16(
                    false, a.v, false, b[t].v, (short)0, acc[t], false, false);
        }
        __syncthreads();         // chunk c fully consumed before buffer reuse
    }

    // Epilogue: max over this wave's 16 proxy rows per column.
    // C/D layout: VGPR r holds M=r (lanes 0-15) / M=r+8 (lanes 16-31), N = lane&15.
    #pragma unroll
    for (int t = 0; t < 4; ++t) {
        float v = acc[t][0];
        #pragma unroll
        for (int r = 1; r < 8; ++r) v = fmaxf(v, acc[t][r]);
        v = fmaxf(v, __shfl_xor(v, 16, 32));           // merge lane halves (M 0-7 vs 8-15)
        if (lane < 16) part[mh][nq * 64 + t * 16 + m] = v;
    }
    __syncthreads();

    out[(size_t)cls * B_N + bbase + tid] = fmaxf(part[0][tid], part[1][tid]);
}

// ---------------------------------------------------------------------------
extern "C" void kernel_launch(void* const* d_in, const int* in_sizes, int n_in,
                              void* d_out, int out_size, void* d_ws, size_t ws_size,
                              hipStream_t stream) {
    const float* data = (const float*)d_in[0];   // [4096, 512]
    const float* w1   = (const float*)d_in[1];   // [64000, 512]
    // d_in[2] segment_ids: class = idx/32, contiguous blocks -> not needed on device.
    (void)in_sizes; (void)n_in; (void)out_size; (void)ws_size;

    uint16_t* d_bf = (uint16_t*)d_ws;                       //  4 MB
    uint16_t* w_bf = d_bf + (size_t)B_N * E_K;              // 64 MB
    float*    out  = (float*)d_out;                         // [2000, 4096]

    norm_bf16_kernel<<<B_N,    256, 0, stream>>>(data, d_bf);
    norm_bf16_kernel<<<P_ROWS, 256, 0, stream>>>(w1,   w_bf);

    dim3 grid(B_N / 256, C_CLS);   // (16, 2000)
    proxy_gemm_max_kernel<<<grid, 256, 0, stream>>>(w_bf, d_bf, out);
}
